// FlashAttention_37237366456747
// MI455X (gfx1250) — matrix operand order repaired and checked by reading, not compiled, not run on hardware
//
#include <hip/hip_runtime.h>
#include <hip/hip_bf16.h>
#include <math.h>

// ---------------------------------------------------------------------------
// Types
// ---------------------------------------------------------------------------
typedef __bf16 bf16;
typedef __attribute__((ext_vector_type(16))) __bf16 v16bf;
typedef __attribute__((ext_vector_type(8)))  float  v8f;

union FragU { v16bf v; uint4 q[2]; };

__device__ inline v8f wmma_bf16(v16bf a, v16bf b, v8f c) {
  return __builtin_amdgcn_wmma_f32_16x16x32_bf16(
      /*neg_a=*/false, a, /*neg_b=*/false, b,
      /*c_mod=*/(short)0, c, /*reuse_a=*/false, /*reuse_b=*/false);
}

__device__ inline bf16 f2bf(float f) {
  unsigned u = __builtin_bit_cast(unsigned, f);
  u += 0x7FFFu + ((u >> 16) & 1u);  // round-to-nearest-even
  unsigned short h = (unsigned short)(u >> 16);
  return __builtin_bit_cast(bf16, h);
}

// ---------------------------------------------------------------------------
// Async global->LDS copy (CDNA5 GLOBAL_LOAD_ASYNC_TO_LDS_B128, ASYNCcnt).
// lds_off = wave-relative LDS byte offset (low 32 bits of generic pointer).
// ---------------------------------------------------------------------------
__device__ inline void async_b128(unsigned lds_off, const void* gaddr) {
  asm volatile("global_load_async_to_lds_b128 %0, %1, off"
               :: "v"(lds_off), "v"((unsigned long long)(size_t)gaddr)
               : "memory");
}

__device__ inline void wait_async0() {
#if __has_builtin(__builtin_amdgcn_s_wait_asynccnt)
  __builtin_amdgcn_s_wait_asynccnt(0);
#else
  asm volatile("s_wait_asynccnt 0x0" ::: "memory");
#endif
}

__device__ inline unsigned lds_off_of(const void* p) {
  return (unsigned)(size_t)p;  // low 32 bits of the generic address are the LDS offset
}

// ---------------------------------------------------------------------------
// WMMA fragment loaders (contiguous k order k = 16*(lane/16) + s for A and B, wave32)
// ---------------------------------------------------------------------------
// A-matrix 16x32 bf16: low lanes K 0-15, high lanes K 16-31.
__device__ inline v16bf load_a_frag(const bf16* base, int row_stride) {
  int lane = threadIdx.x & 31;
  int koff = (lane >> 4) << 4;
  const bf16* pr = base + (lane & 15) * row_stride;
  FragU f;
  f.q[0] = *reinterpret_cast<const uint4*>(pr + koff);
  f.q[1] = *reinterpret_cast<const uint4*>(pr + koff + 8);
  return f.v;
}

// B-matrix 32x16 bf16: low lanes hold K0..15 of their column, high K16..31.
// Source rows are B columns (K contiguous).
__device__ inline v16bf load_b_frag(const bf16* base, int col_stride) {
  int lane = threadIdx.x & 31;
  const bf16* pc = base + (lane & 15) * col_stride + ((lane >> 4) << 4);
  FragU f;
  f.q[0] = *reinterpret_cast<const uint4*>(pc);
  f.q[1] = *reinterpret_cast<const uint4*>(pc + 8);
  return f.v;
}

// B fragment from a row-major KxN tile (columns strided): used for V.
__device__ inline v16bf load_b_frag_strided(const bf16* base, int k_stride) {
  int lane = threadIdx.x & 31;
  int kbase = (lane >> 4) << 4;
  const bf16* pc = base + (lane & 15);
  v16bf f;
#pragma unroll
  for (int e = 0; e < 16; ++e) f[e] = pc[(kbase + e) * k_stride];
  return f;
}

// ---------------------------------------------------------------------------
// f32 -> bf16 conversion
// ---------------------------------------------------------------------------
__global__ void cvt_kernel(const float* __restrict__ src, bf16* __restrict__ dst, int n) {
  int i = blockIdx.x * blockDim.x + threadIdx.x;
  if (i < n) dst[i] = f2bf(src[i]);
}

// ---------------------------------------------------------------------------
// GEMM: C[M,N] = A[M,K] @ B[N,K]^T  (A,B row-major bf16, C f32)
// block tile 128x128, 8 waves (2x4), wave tile 64x32, K panel 64, async LDS.
// ---------------------------------------------------------------------------
#define GLDA 72  // LDS row stride (64 + 8 pad), bf16 elements

__global__ void __launch_bounds__(256)
gemm_bf16_kernel(const bf16* __restrict__ A, const bf16* __restrict__ Bm,
                 float* __restrict__ C, int M, int N, int K) {
  __shared__ bf16 As[128 * GLDA];
  __shared__ bf16 Bs[128 * GLDA];

  int tid  = threadIdx.x;
  int lane = tid & 31;
  int wave = tid >> 5;
  int wm0  = (wave >> 2) * 64;
  int wn0  = (wave & 3) * 32;
  int m0   = blockIdx.y * 128;
  int n0   = blockIdx.x * 128;
  int half = lane >> 4;
  int coln = lane & 15;

  v8f acc[4][2];
  v8f z8 = {0.f, 0.f, 0.f, 0.f, 0.f, 0.f, 0.f, 0.f};
#pragma unroll
  for (int mi = 0; mi < 4; ++mi)
#pragma unroll
    for (int ni = 0; ni < 2; ++ni) acc[mi][ni] = z8;

  // staging: 2 threads/row, 32 elems (64B) each
  int lrow = tid >> 1;
  int lcol = (tid & 1) << 5;
  const bf16* gA = A  + (size_t)(m0 + lrow) * K + lcol;
  const bf16* gB = Bm + (size_t)(n0 + lrow) * K + lcol;
  unsigned sA = lds_off_of(&As[lrow * GLDA + lcol]);
  unsigned sB = lds_off_of(&Bs[lrow * GLDA + lcol]);

  for (int k0 = 0; k0 < K; k0 += 64) {
    __syncthreads();
#pragma unroll
    for (int q = 0; q < 4; ++q) {
      async_b128(sA + q * 16, gA + k0 + q * 8);
      async_b128(sB + q * 16, gB + k0 + q * 8);
    }
    wait_async0();
    __syncthreads();

#pragma unroll
    for (int ks = 0; ks < 2; ++ks) {
      v16bf b0 = load_b_frag(&Bs[wn0 * GLDA + ks * 32], GLDA);
      v16bf b1 = load_b_frag(&Bs[(wn0 + 16) * GLDA + ks * 32], GLDA);
#pragma unroll
      for (int mi = 0; mi < 4; ++mi) {
        v16bf a = load_a_frag(&As[(wm0 + mi * 16) * GLDA + ks * 32], GLDA);
        acc[mi][0] = wmma_bf16(a, b0, acc[mi][0]);
        acc[mi][1] = wmma_bf16(a, b1, acc[mi][1]);
      }
    }
  }

  // epilogue: pointer walk, immediate offsets for the two N tiles
#pragma unroll
  for (int mi = 0; mi < 4; ++mi) {
    float* rp = C + (size_t)(m0 + wm0 + mi * 16 + half * 8) * N + (n0 + wn0 + coln);
#pragma unroll
    for (int r = 0; r < 8; ++r) {
      rp[0]  = acc[mi][0][r];
      rp[16] = acc[mi][1][r];
      rp += N;
    }
  }
}

// ---------------------------------------------------------------------------
// RMSNorm (over hd=64) + RoPE, src [B*L, Hh*64] f32 -> dst [B, Hh, 2048, 64] bf16
// one wave per (token, head); lane owns elems (lane, lane+32) = rotate pair.
// ---------------------------------------------------------------------------
__global__ void norm_rope_kernel(const float* __restrict__ src, const float* __restrict__ w,
                                 bf16* __restrict__ dst, int Hh, float oscale) {
  int gid  = blockIdx.x * blockDim.x + threadIdx.x;
  int wid  = gid >> 5;
  int lane = gid & 31;
  int t  = wid / Hh;           // token = b*2048 + l
  int hh = wid - t * Hh;
  int l  = t & 2047;
  int b  = t >> 11;

  size_t ib = (size_t)t * (Hh * 64) + hh * 64;
  float x1 = src[ib + lane];
  float x2 = src[ib + lane + 32];

  float ss = x1 * x1 + x2 * x2;
#pragma unroll
  for (int off = 16; off > 0; off >>= 1) ss += __shfl_xor(ss, off, 32);
  float inv = rsqrtf(ss * (1.0f / 64.0f) + 1e-6f);
  float n1 = x1 * inv * w[lane];
  float n2 = x2 * inv * w[lane + 32];

  // inv_freq = 10000^(-lane/32) = exp(-lane * ln(10000)/32)
  float fr = (float)l * __expf(-0.28782313662425576f * (float)lane);
  float c, s;
  __sincosf(fr, &s, &c);
  float o1 = (n1 * c - n2 * s) * oscale;
  float o2 = (n2 * c + n1 * s) * oscale;

  size_t ob = ((size_t)(b * Hh + hh) * 2048 + l) * 64;
  dst[ob + lane]      = f2bf(o1);
  dst[ob + lane + 32] = f2bf(o2);
}

// V: [B*L, 8*64] f32 -> [B, 8, 2048, 64] bf16
__global__ void vcvt_kernel(const float* __restrict__ src, bf16* __restrict__ dst) {
  int i = blockIdx.x * blockDim.x + threadIdx.x;  // over 2*2048*512
  int d = i & 63;
  int r = i >> 6;
  int h = r & 7;
  int t = r >> 3;
  int l = t & 2047, b = t >> 11;
  dst[((size_t)(b * 8 + h) * 2048 + l) * 64 + d] = f2bf(src[i]);
}

// ---------------------------------------------------------------------------
// Flash attention (non-causal, GQA 4:1), hd=64, L=2048.
// grid (L/64, H=32, B=2), 128 threads (4 waves); wave owns 16 query rows.
// Q/K in [B, heads, L, 64] bf16 (Q pre-scaled by 1/sqrt(64)).
// Out: [B, L, H, 64] bf16.
// ---------------------------------------------------------------------------
#define ALDK 72  // LDS stride (64 + 8 pad)

__global__ void __launch_bounds__(128)
attn_kernel(const bf16* __restrict__ Q, const bf16* __restrict__ Kt,
            const bf16* __restrict__ Vt, bf16* __restrict__ Oout) {
  __shared__ bf16 Ks[64 * ALDK];
  __shared__ bf16 Vs[64 * ALDK];
  __shared__ bf16 Ps[4 * 16 * ALDK];  // per-wave 16x64 P strip

  int tid = threadIdx.x, lane = tid & 31, wave = tid >> 5;
  int half = lane >> 4, coln = lane & 15;
  int qt = blockIdx.x, h = blockIdx.y, b = blockIdx.z;
  int g = h >> 2;  // kv head

  const bf16* Qh = Q  + (size_t)(b * 32 + h) * 2048 * 64;
  const bf16* Kh = Kt + (size_t)(b * 8 + g) * 2048 * 64;
  const bf16* Vh = Vt + (size_t)(b * 8 + g) * 2048 * 64;

  int qrow0 = qt * 64 + wave * 16;

  // Q fragments stay in VGPRs for the whole KV sweep
  v16bf aq[2];
  aq[0] = load_a_frag(Qh + (size_t)qrow0 * 64, 64);
  aq[1] = load_a_frag(Qh + (size_t)qrow0 * 64 + 32, 64);

  v8f z8 = {0.f, 0.f, 0.f, 0.f, 0.f, 0.f, 0.f, 0.f};
  v8f ao[4];
#pragma unroll
  for (int ni = 0; ni < 4; ++ni) ao[ni] = z8;
  float mst[8], lst[8];
#pragma unroll
  for (int r = 0; r < 8; ++r) { mst[r] = -3.0e38f; lst[r] = 0.f; }

  bf16* Pw = &Ps[wave * 16 * ALDK];

  // staging addresses: 2 threads/row, 32 elems each
  int sr  = tid >> 1;
  int scb = (tid & 1) << 5;
  unsigned sK = lds_off_of(&Ks[sr * ALDK + scb]);
  unsigned sV = lds_off_of(&Vs[sr * ALDK + scb]);
  const bf16* gK = Kh + (size_t)sr * 64 + scb;
  const bf16* gV = Vh + (size_t)sr * 64 + scb;

  for (int j = 0; j < 2048 / 64; ++j) {
    __syncthreads();
#pragma unroll
    for (int q = 0; q < 4; ++q) {
      async_b128(sK + q * 16, gK + j * 64 * 64 + q * 8);
      async_b128(sV + q * 16, gV + j * 64 * 64 + q * 8);
    }
    wait_async0();
    __syncthreads();

    // S strip = Q(16x64) @ K_tile^T (64x64)  -> 4 accumulators of 16x16
    v8f s4[4];
#pragma unroll
    for (int ni = 0; ni < 4; ++ni) {
      v8f sA = z8;
      sA = wmma_bf16(aq[0], load_b_frag(&Ks[(ni * 16) * ALDK], ALDK), sA);
      sA = wmma_bf16(aq[1], load_b_frag(&Ks[(ni * 16) * ALDK + 32], ALDK), sA);
      s4[ni] = sA;
    }

    // online softmax: row stats live per lane-half (16 lanes hold 16 columns)
    float mnew[8], alpha[8];
#pragma unroll
    for (int r = 0; r < 8; ++r) {
      float mx = fmaxf(fmaxf(s4[0][r], s4[1][r]), fmaxf(s4[2][r], s4[3][r]));
#pragma unroll
      for (int off = 1; off < 16; off <<= 1) mx = fmaxf(mx, __shfl_xor(mx, off, 32));
      mnew[r]  = fmaxf(mst[r], mx);
      alpha[r] = __expf(mst[r] - mnew[r]);
      mst[r]   = mnew[r];
    }

#pragma unroll
    for (int ni = 0; ni < 4; ++ni) {
#pragma unroll
      for (int r = 0; r < 8; ++r) {
        float pv = __expf(s4[ni][r] - mnew[r]);
        s4[ni][r] = pv;
        Pw[(half * 8 + r) * ALDK + ni * 16 + coln] = f2bf(pv);
      }
    }

#pragma unroll
    for (int r = 0; r < 8; ++r) {
      float su = s4[0][r] + s4[1][r] + s4[2][r] + s4[3][r];
#pragma unroll
      for (int off = 1; off < 16; off <<= 1) su += __shfl_xor(su, off, 32);
      lst[r] = lst[r] * alpha[r] + su;
#pragma unroll
      for (int ni = 0; ni < 4; ++ni) ao[ni][r] = ao[ni][r] * alpha[r];
    }

    // O += P(16x64) @ V_tile(64x64)   (same-wave LDS ops are in order)
    v16bf ap[2];
    ap[0] = load_a_frag(Pw, ALDK);
    ap[1] = load_a_frag(Pw + 32, ALDK);
#pragma unroll
    for (int ni = 0; ni < 4; ++ni) {
      ao[ni] = wmma_bf16(ap[0], load_b_frag_strided(&Vs[ni * 16], ALDK), ao[ni]);
      ao[ni] = wmma_bf16(ap[1], load_b_frag_strided(&Vs[32 * ALDK + ni * 16], ALDK), ao[ni]);
    }
  }

  // finalize: multiply by reciprocal row sums, pointer-walk the stores
  bf16* op = Oout + ((size_t)(b * 2048 + qrow0 + half * 8) * 32 + h) * 64 + coln;
#pragma unroll
  for (int r = 0; r < 8; ++r) {
    float rl = 1.0f / lst[r];
#pragma unroll
    for (int ni = 0; ni < 4; ++ni)
      op[ni * 16] = f2bf(ao[ni][r] * rl);
    op += 32 * 64;  // next query row in [B, L, H, 64]
  }
}

// ---------------------------------------------------------------------------
// Host-side pipeline
// ---------------------------------------------------------------------------
extern "C" void kernel_launch(void* const* d_in, const int* in_sizes, int n_in,
                              void* d_out, int out_size, void* d_ws, size_t ws_size,
                              hipStream_t stream) {
  const float* x   = (const float*)d_in[0];
  const float* wq  = (const float*)d_in[1];
  const float* wk  = (const float*)d_in[2];
  const float* wv  = (const float*)d_in[3];
  const float* wo  = (const float*)d_in[4];
  const float* qnw = (const float*)d_in[5];
  const float* knw = (const float*)d_in[6];
  float* out = (float*)d_out;

  constexpr int B = 2, L = 2048, D = 2048, H = 32, HKV = 8;
  constexpr int NT  = B * L;       // 4096 tokens
  constexpr int DKV = HKV * 64;    // 512

  char* p = (char*)d_ws;
  auto carve = [&](size_t n) { char* r = p; p += (n + 255) & ~(size_t)255; return r; };
  bf16*  xb  = (bf16*) carve((size_t)NT * D * 2);
  bf16*  wqb = (bf16*) carve((size_t)D * D * 2);
  bf16*  wkb = (bf16*) carve((size_t)DKV * D * 2);
  bf16*  wvb = (bf16*) carve((size_t)DKV * D * 2);
  bf16*  wob = (bf16*) carve((size_t)D * D * 2);
  float* qf  = (float*)carve((size_t)NT * D * 4);
  float* kf  = (float*)carve((size_t)NT * DKV * 4);
  float* vf  = (float*)carve((size_t)NT * DKV * 4);
  bf16*  qb  = (bf16*) carve((size_t)NT * D * 2);
  bf16*  kb  = (bf16*) carve((size_t)NT * DKV * 2);
  bf16*  vb  = (bf16*) carve((size_t)NT * DKV * 2);
  bf16*  ab  = (bf16*) carve((size_t)NT * D * 2);

  auto cvt = [&](const float* s, bf16* d, int n) {
    cvt_kernel<<<(n + 255) / 256, 256, 0, stream>>>(s, d, n);
  };
  cvt(x,  xb,  NT * D);
  cvt(wq, wqb, D * D);
  cvt(wk, wkb, DKV * D);
  cvt(wv, wvb, DKV * D);
  cvt(wo, wob, D * D);

  // projections
  gemm_bf16_kernel<<<dim3(D / 128,   NT / 128), 256, 0, stream>>>(xb, wqb, qf, NT, D,   D);
  gemm_bf16_kernel<<<dim3(DKV / 128, NT / 128), 256, 0, stream>>>(xb, wkb, kf, NT, DKV, D);
  gemm_bf16_kernel<<<dim3(DKV / 128, NT / 128), 256, 0, stream>>>(xb, wvb, vf, NT, DKV, D);

  // RMSNorm + RoPE (fold 1/sqrt(hd) into Q); layout to [B, heads, L, 64] bf16
  norm_rope_kernel<<<(NT * H * 32) / 256,   256, 0, stream>>>(qf, qnw, qb, H,   0.125f);
  norm_rope_kernel<<<(NT * HKV * 32) / 256, 256, 0, stream>>>(kf, knw, kb, HKV, 1.0f);
  vcvt_kernel<<<(NT * DKV) / 256, 256, 0, stream>>>(vf, vb);

  // attention
  attn_kernel<<<dim3(L / 64, H, B), 128, 0, stream>>>(qb, kb, vb, ab);

  // output projection -> fp32 d_out
  gemm_bf16_kernel<<<dim3(D / 128, NT / 128), 256, 0, stream>>>(ab, wob, out, NT, D, D);
}
